// unit_gcn_sparse_32006096290313
// MI455X (gfx1250) — matrix operand order repaired and checked by reading, not compiled, not run on hardware
//
#include <hip/hip_runtime.h>
#include <hip/hip_bf16.h>

typedef __attribute__((ext_vector_type(16))) __bf16 v16bf;
typedef __attribute__((ext_vector_type(8)))  __bf16 v8bf;
typedef __attribute__((ext_vector_type(8)))  float  v8f;

#define N_    32
#define CIN_  256
#define COUT_ 256
#define T_    128
#define V_    25
#define K_    3
#define TV_   (T_*V_)      // 3200
#define TC_   16           // t's per workgroup
#define MT_   (TC_*V_)     // 400 columns per workgroup
#define KC_   32           // cin chunk per K-step
#define WLP_  40           // Wl pitch (bf16 units), 80B rows -> 16B aligned frags
#define XLP_  40           // Xl pitch (bf16 units)
#define YP_   400          // y1 tile pitch (f32 units)
#define CNT_  102400.0f    // N*T*V per channel
#define BN_EPS 1e-5f

union ABFrag { v16bf v; v8bf h[2]; };

__device__ __forceinline__ unsigned short f2bf(float f) {
    unsigned u = __float_as_uint(f);
    u += 0x7FFFu + ((u >> 16) & 1u);      // round-to-nearest-even
    return (unsigned short)(u >> 16);
}

// Kernel 1: fused 1x1-conv (bf16 WMMA, f32 accum) + graph contraction + BN partials.
// Per-wave tile block: 3 o-tiles x 4 m-tiles with fragments held in registers:
// 14 ds_load_b128 feed 12 WMMAs per K-step (vs 4 loads per WMMA before).
__global__ __launch_bounds__(256) void gcn_fused(
    const float* __restrict__ x, const float* __restrict__ Ag,
    const float* __restrict__ Wg, const float* __restrict__ bg,
    float* __restrict__ y2, float* __restrict__ P1, float* __restrict__ P2)
{
    __shared__ __align__(16) unsigned short Wl[48 * WLP_];   //  3,840 B
    __shared__ __align__(16) unsigned short Xl[MT_ * XLP_];  // 32,000 B (transposed x tile)
    __shared__ float Ysh[48 * YP_];                          // 76,800 B (y1 tile, f32)
    __shared__ float bl[48];
    __shared__ float red1[256];
    __shared__ float red2[256];

    const int tid = threadIdx.x;
    const int bid = blockIdx.x;
    const int n   = bid >> 7;          // 0..31
    const int cc  = (bid >> 3) & 15;   // channel chunk
    const int tc  = bid & 7;           // t chunk
    const int c0  = cc * 16;
    const int t0  = tc * TC_;

    if (tid < 48) bl[tid] = bg[(tid >> 4) * COUT_ + c0 + (tid & 15)];

    const int wave  = tid >> 5;
    const int lane  = tid & 31;
    const int laneM = lane & 15;
    const int laneH = lane >> 4;       // 0: lanes 0-15, 1: lanes 16-31
    const int miBase = wave * 3;       // waves cover mi 0..24 with overlap (dup writes identical)

    v8f acc[12];
#pragma unroll
    for (int i = 0; i < 12; ++i) acc[i] = {};

    const float* xbase = x + (size_t)n * CIN_ * TV_ + (size_t)t0 * V_;
    unsigned* Wl32 = (unsigned*)Wl;
    unsigned* Xl32 = (unsigned*)Xl;

    for (int kk = 0; kk < CIN_ / KC_; ++kk) {
        const int cin0 = kk * KC_;
        __syncthreads();   // previous WMMA reads complete before overwrite
        // W chunk: 48 rows (k-stacked 16-channel groups) x 32 cin; float2 load, packed bf16x2 store
        {
            const int p  = tid & 15;      // cin pair
            const int rb = tid >> 4;      // 0..15
#pragma unroll
            for (int i = 0; i < 3; ++i) {
                int r = rb + i * 16;
                int o = (r >> 4) * COUT_ + c0 + (r & 15);
                float2 wv = *(const float2*)(Wg + (size_t)o * CIN_ + cin0 + 2 * p);
                Wl32[r * (WLP_ / 2) + p] = (unsigned)f2bf(wv.x) | ((unsigned)f2bf(wv.y) << 16);
            }
        }
        // x chunk: 32 cin x 400 m, transposed Xl[m][cin]; pack cin-pairs into u32 stores
        {
#pragma unroll
            for (int i = 0; i < 25; ++i) {
                int flat = tid + i * 256;           // 0..6399
                int ck2  = flat / MT_;              // 0..15 (cin pair)
                int m    = flat - ck2 * MT_;
                const float* xp = xbase + (size_t)(cin0 + 2 * ck2) * TV_ + m;
                float lo = xp[0];
                float hi = xp[TV_];
                Xl32[m * (XLP_ / 2) + ck2] = (unsigned)f2bf(lo) | ((unsigned)f2bf(hi) << 16);
            }
        }
        __syncthreads();
        // Hoist fragments into registers: 3 A-frags + 4 B-frags -> 12 WMMAs
        ABFrag af[3];
#pragma unroll
        for (int oi = 0; oi < 3; ++oi) {
            // A (16x32 bf16): per lane two 16B halves: K 0-7/16-23 (lo lanes), 8-15/24-31 (hi)
            af[oi].h[0] = *(const v8bf*)(Wl + (oi * 16 + laneM) * WLP_ + laneH * 8);
            af[oi].h[1] = *(const v8bf*)(Wl + (oi * 16 + laneM) * WLP_ + 16 + laneH * 8);
        }
        ABFrag bf[4];
#pragma unroll
        for (int j = 0; j < 4; ++j) {
            int mi = miBase + j;                    // <= 24 always
            // B (32x16 bf16): lane holds column N=laneM, K 0-7/16-23 (lo lanes), 8-15/24-31 (hi)
            bf[j].h[0] = *(const v8bf*)(Xl + (mi * 16 + laneM) * XLP_ + laneH * 8);
            bf[j].h[1] = *(const v8bf*)(Xl + (mi * 16 + laneM) * XLP_ + 16 + laneH * 8);
        }
#pragma unroll
        for (int j = 0; j < 4; ++j)
#pragma unroll
            for (int oi = 0; oi < 3; ++oi)
                acc[j * 3 + oi] = __builtin_amdgcn_wmma_f32_16x16x32_bf16(
                    false, af[oi].v, false, bf[j].v, (short)0, acc[j * 3 + oi], false, false);
    }
    __syncthreads();
    // Spill accumulators (+bias) to LDS y1 tile: D layout M = r + 8*laneH, N = laneM
#pragma unroll
    for (int j = 0; j < 4; ++j) {
        const int mi = miBase + j;
#pragma unroll
        for (int oi = 0; oi < 3; ++oi) {
#pragma unroll
            for (int r = 0; r < 8; ++r) {
                int row = oi * 16 + r + laneH * 8;
                Ysh[row * YP_ + mi * 16 + laneM] = acc[j * 3 + oi][r] + bl[row];
            }
        }
    }
    __syncthreads();

    // Stage 2: out[c,t,w] = sum_k sum_v y1[k,c,t,v] * A[k,v,w]
    // A address is lane-uniform -> compiler scalarizes to SMEM broadcast loads.
    const int ci = tid >> 4;    // channel within chunk
    const int tt = tid & 15;    // t within chunk
    float yv[K_ * V_];
#pragma unroll
    for (int k = 0; k < K_; ++k)
#pragma unroll
        for (int v = 0; v < V_; ++v)
            yv[k * V_ + v] = Ysh[(k * 16 + ci) * YP_ + tt * V_ + v];

    float s1 = 0.f, s2 = 0.f;
    float* outp = y2 + (size_t)n * COUT_ * TV_ + (size_t)(c0 + ci) * TV_ + (size_t)(t0 + tt) * V_;
#pragma unroll 1
    for (int w = 0; w < V_; ++w) {
        float s = 0.f;
#pragma unroll
        for (int k = 0; k < K_; ++k)
#pragma unroll
            for (int v = 0; v < V_; ++v)
                s = fmaf(yv[k * V_ + v], Ag[(k * V_ + v) * V_ + w], s);
        outp[w] = s;
        s1 += s;
        s2 = fmaf(s, s, s2);
    }
    // Deterministic fixed-order reduction across the 16 t-threads per channel
    red1[tid] = s1; red2[tid] = s2;
    __syncthreads();
    if (tt == 0) {
        float a1 = 0.f, a2 = 0.f;
#pragma unroll
        for (int j = 0; j < 16; ++j) { a1 += red1[ci * 16 + j]; a2 += red2[ci * 16 + j]; }
        const int c = c0 + ci;
        P1[c * 256 + n * 8 + tc] = a1;   // unique slot per WG -> deterministic
        P2[c * 256 + n * 8 + tc] = a2;
    }
}

// Kernel 2: fold 256 partials per channel (fixed order) -> scale/shift
__global__ __launch_bounds__(256) void bn_stats(
    const float* __restrict__ P1, const float* __restrict__ P2,
    const float* __restrict__ gamma, const float* __restrict__ beta,
    float* __restrict__ SS, float* __restrict__ SH)
{
    const int c = threadIdx.x;
    float s1 = 0.f, s2 = 0.f;
    for (int j = 0; j < 256; ++j) { s1 += P1[c * 256 + j]; s2 += P2[c * 256 + j]; }
    const float mu  = s1 * (1.0f / CNT_);
    const float var = s2 * (1.0f / CNT_) - mu * mu;
    const float inv = rsqrtf(var + BN_EPS);
    const float sc  = gamma[c] * inv;
    SS[c] = sc;
    SH[c] = beta[c] - mu * sc;
}

// Kernel 3: in-place BN apply + ReLU (float4)
__global__ __launch_bounds__(256) void bn_apply(
    float* __restrict__ y, const float* __restrict__ SS, const float* __restrict__ SH)
{
    const int idx4 = blockIdx.x * 256 + threadIdx.x;   // 3200 % 4 == 0 -> same channel
    const int c = (idx4 / 800) & 255;
    float4 v = ((float4*)y)[idx4];
    const float sc = SS[c], sh = SH[c];
    v.x = fmaxf(fmaf(v.x, sc, sh), 0.f);
    v.y = fmaxf(fmaf(v.y, sc, sh), 0.f);
    v.z = fmaxf(fmaf(v.z, sc, sh), 0.f);
    v.w = fmaxf(fmaf(v.w, sc, sh), 0.f);
    ((float4*)y)[idx4] = v;
}

extern "C" void kernel_launch(void* const* d_in, const int* in_sizes, int n_in,
                              void* d_out, int out_size, void* d_ws, size_t ws_size,
                              hipStream_t stream) {
    const float* x     = (const float*)d_in[0];
    const float* A     = (const float*)d_in[1];
    const float* W     = (const float*)d_in[2];
    const float* b     = (const float*)d_in[3];
    const float* gamma = (const float*)d_in[4];
    const float* beta  = (const float*)d_in[5];
    float* out = (float*)d_out;
    float* ws  = (float*)d_ws;

    float* P1 = ws;                 // [256][256]
    float* P2 = ws + 65536;         // [256][256]
    float* SS = ws + 131072;        // [256]
    float* SH = ws + 131328;        // [256]

    gcn_fused<<<dim3(N_ * 16 * 8), dim3(256), 0, stream>>>(x, A, W, b, out, P1, P2);
    bn_stats<<<dim3(1), dim3(256), 0, stream>>>(P1, P2, gamma, beta, SS, SH);
    bn_apply<<<dim3((N_ * COUT_ * TV_) / 1024), dim3(256), 0, stream>>>(out, SS, SH);
}
